// CascadeGNN_67800353734945
// MI455X (gfx1250) — compile-verified, matrix operand order checked by reading, not run on hardware
//
#include <hip/hip_runtime.h>
#include <hip/hip_bf16.h>
#include <math.h>

typedef __attribute__((ext_vector_type(16))) _Float16 v16h;
typedef __attribute__((ext_vector_type(8)))  float    v8f;

#define NN   100000
#define EE   1600000
#define GG   128
#define DIN  8
#define HH   64
#define CC   2

// -------------------- utility --------------------
__global__ void k_zero(float* __restrict__ p, int n) {
  int i = blockIdx.x * blockDim.x + threadIdx.x;
  if (i < n) p[i] = 0.0f;
}

// deg[dst] += 1 per edge (target array is 0.4 MB -> L2-resident atomics)
__global__ void k_deg(const int* __restrict__ dst, float* __restrict__ deg, int e) {
  int i = blockIdx.x * blockDim.x + threadIdx.x;
  if (i < e) atomicAdd(&deg[dst[i]], 1.0f);
}

// dis = rsqrt(deg + 1) in place
__global__ void k_dis(float* __restrict__ d, int n) {
  int i = blockIdx.x * blockDim.x + threadIdx.x;
  if (i < n) d[i] = rsqrtf(d[i] + 1.0f);
}

// h = relu(x @ W_emb + b_emb); one thread per (node, channel), K=8 dot in registers
__global__ void k_embed(const float* __restrict__ x, const float* __restrict__ W,
                        const float* __restrict__ b, float* __restrict__ h, int n) {
  int gid = blockIdx.x * blockDim.x + threadIdx.x;
  if (gid >= n * HH) return;
  int node = gid >> 6, c = gid & 63;
  float acc = b[c];
#pragma unroll
  for (int k = 0; k < DIN; ++k) acc = fmaf(x[node * DIN + k], W[k * HH + c], acc);
  h[gid] = fmaxf(acc, 0.0f);
}

// -------------------- WMMA GEMM: xt[N,64] = relu(h)[N,64] @ W[64,64] ------------
// W is staged transposed + f16 in LDS (8 KB) once per block; B fragments are then
// two contiguous 16B LDS reads each (ds_load_b128). One wave per 16-node tile;
// 2 K-steps x 4 column tiles = 8 v_wmma per wave. ReLU applied on A-load
// (deferred from the previous layer's epilogue; idempotent for layer 0).
__global__ void k_gemm_wmma(const float* __restrict__ h, const float* __restrict__ W,
                            float* __restrict__ xt, int nTiles) {
  __shared__ __align__(16) _Float16 Wt[HH * HH];   // Wt[col*64 + k]
  for (int idx = threadIdx.x; idx < HH * HH; idx += blockDim.x) {
    int k = idx >> 6, col = idx & 63;
    Wt[col * HH + k] = (_Float16)W[idx];
  }
  __syncthreads();

  const int wave = (blockIdx.x * blockDim.x + threadIdx.x) >> 5;
  if (wave >= nTiles) return;                 // wave-uniform: EXEC stays all-ones
  const int lane = threadIdx.x & 31;
  const int r  = lane & 15;                   // row (A) / col (B,C)
  const int hs = lane >> 4;                   // half-select per ISA 16-bit layout
  const int node = wave * 16 + r;
  const float* __restrict__ hrow = h + (size_t)node * HH;

  // Prefetch next A panel (speculative; OOB prefetch silently dropped)
  __builtin_prefetch(h + ((size_t)node + 16) * HH, 0, 0);

  v8f acc[4];
#pragma unroll
  for (int ct = 0; ct < 4; ++ct) acc[ct] = (v8f){0.f,0.f,0.f,0.f,0.f,0.f,0.f,0.f};

#pragma unroll
  for (int kb = 0; kb < 2; ++kb) {
    const int k0 = kb * 32 + hs * 8;          // VGPRs 0..3 -> K = k0..k0+7
    const int k1 = kb * 32 + 16 + hs * 8;     // VGPRs 4..7 -> K = k1..k1+7
    v16h a;
#pragma unroll
    for (int i = 0; i < 8; ++i) {
      a[i]     = (_Float16)fmaxf(hrow[k0 + i], 0.0f);
      a[8 + i] = (_Float16)fmaxf(hrow[k1 + i], 0.0f);
    }
#pragma unroll
    for (int ct = 0; ct < 4; ++ct) {
      const _Float16* __restrict__ wp = &Wt[(ct * 16 + r) * HH];
      v16h bf;
#pragma unroll
      for (int i = 0; i < 8; ++i) {
        bf[i]     = wp[k0 + i];               // 16B-aligned contiguous -> ds_load_b128
        bf[8 + i] = wp[k1 + i];
      }
      acc[ct] = __builtin_amdgcn_wmma_f32_16x16x32_f16(
          /*neg_a=*/false, a, /*neg_b=*/false, bf,
          /*c_mod=*/(short)0, acc[ct], /*reuse_a=*/false, /*reuse_b=*/false);
    }
  }

  // C/D layout: lane (r, hs): element v -> row hs*8+v, col r (within tile)
#pragma unroll
  for (int ct = 0; ct < 4; ++ct) {
#pragma unroll
    for (int v = 0; v < 8; ++v) {
      xt[(size_t)(wave * 16 + hs * 8 + v) * HH + ct * 16 + r] = acc[ct][v];
    }
  }
}

// A = dis^2 * xt + bias   (self-loop term; replaces the zero-fill pass)
__global__ void k_init(float* __restrict__ A, const float* __restrict__ xt,
                       const float* __restrict__ dis, const float* __restrict__ b, int n) {
  int gid = blockIdx.x * blockDim.x + threadIdx.x;
  if (gid >= n * 16) return;
  int node = gid >> 4, c4 = (gid & 15) * 4;
  float dd = dis[node];
  float s = dd * dd;
  const float4 v = *(const float4*)(xt + (size_t)node * HH + c4);
  float4 o;
  o.x = s * v.x + b[c4 + 0];
  o.y = s * v.y + b[c4 + 1];
  o.z = s * v.z + b[c4 + 2];
  o.w = s * v.w + b[c4 + 3];
  *(float4*)(A + (size_t)node * HH + c4) = o;
}

// A[dst] += (dis[src]*dis[dst]) * xt[src]; 16 threads/edge, float4 per thread.
// Target array (25.6 MB) is L2-resident -> atomics stay on-chip.
__global__ void k_scatter(const int* __restrict__ src, const int* __restrict__ dst,
                          const float* __restrict__ dis, const float* __restrict__ xt,
                          float* __restrict__ agg, int e) {
  int gid = blockIdx.x * blockDim.x + threadIdx.x;
  if (gid >= e * 16) return;
  int ed = gid >> 4, c4 = (gid & 15) * 4;
  int s = src[ed], d = dst[ed];
  float nrm = dis[s] * dis[d];
  const float4 v = *(const float4*)(xt + (size_t)s * HH + c4);
  float* ap = agg + (size_t)d * HH + c4;
  atomicAdd(ap + 0, nrm * v.x);
  atomicAdd(ap + 1, nrm * v.y);
  atomicAdd(ap + 2, nrm * v.z);
  atomicAdd(ap + 3, nrm * v.w);
}

// per-graph sum / count / max over relu(h) (max via uint atomics: values >= 0)
__global__ void k_pool(const float* __restrict__ h, const int* __restrict__ batch,
                       float* __restrict__ sum, unsigned int* __restrict__ mx,
                       float* __restrict__ cnt, int n) {
  int gid = blockIdx.x * blockDim.x + threadIdx.x;
  if (gid >= n * 16) return;
  int node = gid >> 4, t = gid & 15, c4 = t * 4;
  int g = batch[node];
  const float4 v = *(const float4*)(h + (size_t)node * HH + c4);
  float r0 = fmaxf(v.x, 0.0f), r1 = fmaxf(v.y, 0.0f);
  float r2 = fmaxf(v.z, 0.0f), r3 = fmaxf(v.w, 0.0f);
  float* sp = sum + g * HH + c4;
  unsigned int* mp = mx + g * HH + c4;
  atomicAdd(sp + 0, r0);  atomicMax(mp + 0, __float_as_uint(r0));
  atomicAdd(sp + 1, r1);  atomicMax(mp + 1, __float_as_uint(r1));
  atomicAdd(sp + 2, r2);  atomicMax(mp + 2, __float_as_uint(r2));
  atomicAdd(sp + 3, r3);  atomicMax(mp + 3, __float_as_uint(r3));
  if (t == 0) atomicAdd(&cnt[g], 1.0f);
}

// hg = [mean|max] @ W_pool + b_pool ; logits = hg @ W_cls + b_cls ; log_softmax
__global__ void k_head(const float* __restrict__ sum, const unsigned int* __restrict__ mx,
                       const float* __restrict__ cnt,
                       const float* __restrict__ Wp, const float* __restrict__ bp,
                       const float* __restrict__ Wc, const float* __restrict__ bc,
                       float* __restrict__ out) {
  int g = blockIdx.x * blockDim.x + threadIdx.x;
  if (g >= GG) return;
  float inv = 1.0f / fmaxf(cnt[g], 1.0f);
  float l0 = bc[0], l1 = bc[1];
  for (int j = 0; j < HH; ++j) {
    float a = bp[j];
#pragma unroll 4
    for (int i = 0; i < HH; ++i) {
      a = fmaf(sum[g * HH + i] * inv,           Wp[i * HH + j],        a);
      a = fmaf(__uint_as_float(mx[g * HH + i]), Wp[(HH + i) * HH + j], a);
    }
    l0 = fmaf(a, Wc[j * CC + 0], l0);
    l1 = fmaf(a, Wc[j * CC + 1], l1);
  }
  float m = fmaxf(l0, l1);
  float lse = m + logf(expf(l0 - m) + expf(l1 - m));
  out[g * CC + 0] = l0 - lse;
  out[g * CC + 1] = l1 - lse;
}

// -------------------- launch --------------------
extern "C" void kernel_launch(void* const* d_in, const int* in_sizes, int n_in,
                              void* d_out, int out_size, void* d_ws, size_t ws_size,
                              hipStream_t stream) {
  const float* x     = (const float*)d_in[0];
  const int*   src   = (const int*)  d_in[1];
  const int*   dst   = (const int*)  d_in[2];
  const int*   batch = (const int*)  d_in[3];
  const float* W_emb = (const float*)d_in[4];
  const float* b_emb = (const float*)d_in[5];
  const float* W_g1  = (const float*)d_in[6];
  const float* b_g1  = (const float*)d_in[7];
  const float* W_g2  = (const float*)d_in[8];
  const float* b_g2  = (const float*)d_in[9];
  const float* W_pl  = (const float*)d_in[10];
  const float* b_pl  = (const float*)d_in[11];
  const float* W_cl  = (const float*)d_in[12];
  const float* b_cl  = (const float*)d_in[13];
  float* out = (float*)d_out;

  // workspace (floats): dis[N] | A[N*H] | B[N*H] | psum[G*H] | pmax[G*H] | pcnt[G]
  float* ws   = (float*)d_ws;
  float* dis  = ws;
  float* A    = dis + NN;
  float* B    = A + (size_t)NN * HH;
  float* psum = B + (size_t)NN * HH;
  float* pmax = psum + GG * HH;
  float* pcnt = pmax + GG * HH;

  const int TB = 256;
  const int NH  = NN * HH;                         // 6.4M
  const int N16 = NN * 16;                         // 1.6M  (float4 grids)
  const int E16 = EE * 16;                         // 25.6M (float4 scatter grid)
  const int nTiles = NN / 16;                      // 6250 (N % 16 == 0)
  dim3 blk(TB);

  // degrees -> dis
  k_zero<<<(NN + TB - 1) / TB, blk, 0, stream>>>(dis, NN);
  k_deg <<<(EE + TB - 1) / TB, blk, 0, stream>>>(dst, dis, EE);
  k_dis <<<(NN + TB - 1) / TB, blk, 0, stream>>>(dis, NN);

  // embedding (already ReLU'd)
  k_embed<<<(NH + TB - 1) / TB, blk, 0, stream>>>(x, W_emb, b_emb, A, NN);

  // GCN layer 1: B = relu(A)@W1 ; A = dis^2*B + b1 ; A[dst] += norm*B[src]
  k_gemm_wmma<<<(nTiles * 32 + TB - 1) / TB, blk, 0, stream>>>(A, W_g1, B, nTiles);
  k_init   <<<(N16 + TB - 1) / TB, blk, 0, stream>>>(A, B, dis, b_g1, NN);
  k_scatter<<<(E16 + TB - 1) / TB, blk, 0, stream>>>(src, dst, dis, B, A, EE);

  // GCN layer 2 (ReLU of layer 1 applied on GEMM A-load)
  k_gemm_wmma<<<(nTiles * 32 + TB - 1) / TB, blk, 0, stream>>>(A, W_g2, B, nTiles);
  k_init   <<<(N16 + TB - 1) / TB, blk, 0, stream>>>(A, B, dis, b_g2, NN);
  k_scatter<<<(E16 + TB - 1) / TB, blk, 0, stream>>>(src, dst, dis, B, A, EE);

  // pooling (ReLU of layer 2 applied on load) + head
  const int poolN = 2 * GG * HH + GG;
  k_zero<<<(poolN + TB - 1) / TB, blk, 0, stream>>>(psum, poolN);
  k_pool<<<(N16 + TB - 1) / TB, blk, 0, stream>>>(A, batch, psum,
                                                  (unsigned int*)pmax, pcnt, NN);
  k_head<<<1, dim3(GG), 0, stream>>>(psum, (const unsigned int*)pmax, pcnt,
                                     W_pl, b_pl, W_cl, b_cl, out);
}
